// SchNetDecoder_82154134438118
// MI455X (gfx1250) — matrix operand order repaired and checked by reading, not compiled, not run on hardware
//
#include <hip/hip_runtime.h>

#define NN 50000
#define EE 800000
#define HH 128
#define GG 50

static constexpr float kCut   = 6.0f;
static constexpr float kPi    = 3.14159265358979323846f;
static constexpr float kShift = 0.69314718055994530942f;
static constexpr float kStep  = kCut / (GG - 1);
static constexpr float kCoeff = -0.5f / (kStep * kStep);
static constexpr float kLog2e = 1.44269504088896340736f;
static constexpr float kLn2   = 0.69314718055994530942f;

typedef __attribute__((ext_vector_type(16))) _Float16 v16h;
typedef __attribute__((ext_vector_type(8)))  _Float16 v8h;
typedef __attribute__((ext_vector_type(8)))  float    v8f;

static __device__ __forceinline__ void sched_fence() {
#if __has_builtin(__builtin_amdgcn_sched_barrier)
  __builtin_amdgcn_sched_barrier(0);  // keep load batch ahead of WMMA chain
#endif
}

// Shifted-softplus in the log2 domain: raw v_exp_f32/v_log_f32 (both log2-
// based), no ln2 fixup chains. Select guards exp2 overflow for large x.
static __device__ __forceinline__ float sspf(float x) {
#if __has_builtin(__builtin_amdgcn_exp2f) && __has_builtin(__builtin_amdgcn_logf)
  const float e2 = __builtin_amdgcn_exp2f(x * kLog2e);
  const float l2 = __builtin_amdgcn_logf(1.0f + e2);
  const float sp = (x > 20.0f) ? x : kLn2 * l2;
#else
  const float sp = (x > 20.0f) ? x : __logf(1.0f + __expf(x));
#endif
  return sp - kShift;
}

static __device__ __forceinline__ v8f wmma_16x16x32(v16h a, v16h b, v8f c) {
  // 8 args: (neg_a, A, neg_b, B, c_mod, C, reuse_a, reuse_b)
  return __builtin_amdgcn_wmma_f32_16x16x32_f16(false, a, false, b, (short)0, c,
                                                false, false);
}

// ---------------------------------------------------------------------------
// v = z @ init_w + init_b   ([N,3]@[3,128])  — K=3, pure VALU
// ---------------------------------------------------------------------------
__global__ void schnet_init_k(const float* __restrict__ z,
                              const float* __restrict__ w,
                              const float* __restrict__ b,
                              float* __restrict__ v) {
  int t = blockIdx.x * blockDim.x + threadIdx.x;
  if (t >= NN * HH) return;
  int n = t >> 7, h = t & 127;
  const float* zr = z + n * 3;
  v[t] = b[h] + zr[0] * w[h] + zr[1] * w[HH + h] + zr[2] * w[2 * HH + h];
}

__global__ void schnet_zero_k(float* __restrict__ p, int n) {
  int t = blockIdx.x * blockDim.x + threadIdx.x;
  if (t < n) p[t] = 0.0f;
}

// ---------------------------------------------------------------------------
// Node GEMM: Out = [residual +] act(A @ W + bias)   A:[nrows,128] W:[128,128]
// One wave per 16-row tile; W transposed to [N][K] f16 in LDS so each lane's
// B fragment is one contiguous 32-byte load. Per output tile: batch-load all
// 4 K-block B fragments (sched-fenced), then 4 back-to-back WMMAs.
// ---------------------------------------------------------------------------
__global__ __launch_bounds__(128) void schnet_node_gemm_k(
    const float* __restrict__ A, const float* __restrict__ W,
    const float* __restrict__ bias, const float* residual, float* Out,
    int nrows, int do_ssp) {
  __shared__ __align__(32) _Float16 ws[HH * HH];  // [N][K] transposed, f16
  const int tid = threadIdx.x;
  for (int idx = tid; idx < HH * HH; idx += 128) {
    int n = idx >> 7, k = idx & 127;
    ws[idx] = (_Float16)W[k * HH + n];
  }
  __syncthreads();

  const int wave = tid >> 5, lane = tid & 31;
  const int m = lane & 15, g = lane >> 4;
  const int tile = blockIdx.x * 4 + wave;
  if (tile * 16 >= nrows) return;  // wave-uniform: EXEC all-ones inside
  const int row = tile * 16 + m;

  // A fragments: lane holds M=m, K = kb*32 + 16g + t
  v16h afrag[4];
#pragma unroll
  for (int kb = 0; kb < 4; ++kb) {
    const float* ap = A + row * HH + kb * 32 + 16 * g;
    v16h a;
#pragma unroll
    for (int t = 0; t < 8; ++t) {
      a[t] = (_Float16)ap[t];
      a[t + 8] = (_Float16)ap[8 + t];
    }
    afrag[kb] = a;
  }

#pragma unroll 1
  for (int nt = 0; nt < 8; ++nt) {
    v16h bfrag[4];
#pragma unroll
    for (int kb = 0; kb < 4; ++kb)
      bfrag[kb] = *(const v16h*)&ws[(nt * 16 + m) * HH + kb * 32 + 16 * g];
    sched_fence();  // all 8 ds_load_b128 in flight before first WMMA
    v8f acc = {};
#pragma unroll
    for (int kb = 0; kb < 4; ++kb) acc = wmma_16x16x32(afrag[kb], bfrag[kb], acc);

    const int ncol = nt * 16 + m;
    const float bv = bias ? bias[ncol] : 0.0f;
#pragma unroll
    for (int r = 0; r < 8; ++r) {  // C layout: row M = r + 8g
      const int orow = tile * 16 + r + 8 * g;
      float x = acc[r] + bv;
      if (do_ssp) x = sspf(x);
      if (residual) x += residual[orow * HH + ncol];
      Out[orow * HH + ncol] = x;
    }
  }
}

// ---------------------------------------------------------------------------
// Fused edge kernel (one interaction layer's message pass):
//   W = (ssp(rbf(trunc(d)) @ m0 + b0) @ m2 + b2) * cosgate(d)
//   agg[i] += vlin[j] * W
// 2 waves/block, one 16-edge tile per wave. RBF built directly into WMMA A
// fragments (K padded 50->64). ssp output staged through LDS to re-shape
// C-layout -> A-layout for the second GEMM. Scatter via global f32 atomics.
// ---------------------------------------------------------------------------
__global__ __launch_bounds__(64) void schnet_edge_k(
    const float* __restrict__ dist, const int* __restrict__ eidx,
    const float* __restrict__ m0w, const float* __restrict__ m0b,
    const float* __restrict__ m2w, const float* __restrict__ m2b,
    const float* __restrict__ vlin, float* __restrict__ agg) {
  __shared__ __align__(32) _Float16 b1[HH * 64];        // m0 [N][Kpad=64] f16
  __shared__ __align__(32) _Float16 b2[HH * HH];        // m2 [N][K=128]  f16
  __shared__ __align__(32) _Float16 htmp[2 * 16 * HH];  // per-wave [M][K] f16
  __shared__ float sb0[HH], sb2[HH];
  __shared__ float cg[2][16];
  __shared__ int jr[2][16], ir[2][16];

  const int tid = threadIdx.x;
  for (int idx = tid; idx < HH * 64; idx += 64) {
    int n = idx >> 6, k = idx & 63;
    b1[idx] = (k < GG) ? (_Float16)m0w[k * HH + n] : (_Float16)0.0f;
  }
  for (int idx = tid; idx < HH * HH; idx += 64) {
    int n = idx >> 7, k = idx & 127;
    b2[idx] = (_Float16)m2w[k * HH + n];
  }
  for (int idx = tid; idx < HH; idx += 64) {
    sb0[idx] = m0b[idx];
    sb2[idx] = m2b[idx];
  }

  const int wave = tid >> 5, lane = tid & 31;
  const int m = lane & 15, g = lane >> 4;
  const int eb = (blockIdx.x * 2 + wave) * 16;  // E = 25000*2*16 exactly

  if (lane < 16) {  // per-tile edge metadata (no barriers inside)
    const int e = eb + lane;
    const float d = dist[e];
    const int jv = eidx[e];         // source
    cg[wave][lane] = 0.5f * (__cosf(d * (kPi / kCut)) + 1.0f);
    jr[wave][lane] = jv;
    ir[wave][lane] = eidx[EE + e];  // target
    // Warm the gather rows (L2-resident vlin table) while the GEMMs run.
    __builtin_prefetch(vlin + (size_t)jv * HH, 0, 1);
    __builtin_prefetch(vlin + (size_t)jv * HH + 64, 0, 1);
  }
  __syncthreads();

  // RBF A fragments: edge row M=m, gaussian index = K
  const float td = truncf(dist[eb + m]);
  v16h a1frag[2];
#pragma unroll
  for (int kb = 0; kb < 2; ++kb) {
    v16h a;
#pragma unroll
    for (int t = 0; t < 16; ++t) {
      const int k = kb * 32 + 16 * g + t;
      float val = 0.0f;
      if (k < GG) {
        const float dd = td - (float)k * kStep;
        val = __expf(kCoeff * dd * dd);
      }
      a[t] = (_Float16)val;
    }
    a1frag[kb] = a;
  }

  _Float16* hw = &htmp[wave * 16 * HH];

  // GEMM1: [16,64]@[64,128] + bias, ssp -> stage to LDS as [M][K]
#pragma unroll 1
  for (int nt = 0; nt < 8; ++nt) {
    v16h bfrag[2];
#pragma unroll
    for (int kb = 0; kb < 2; ++kb)
      bfrag[kb] = *(const v16h*)&b1[(nt * 16 + m) * 64 + kb * 32 + 16 * g];
    sched_fence();
    v8f acc = {};
#pragma unroll
    for (int kb = 0; kb < 2; ++kb)
      acc = wmma_16x16x32(a1frag[kb], bfrag[kb], acc);

    const int ncol = nt * 16 + m;
    const float bv = sb0[ncol];
#pragma unroll
    for (int r = 0; r < 8; ++r)
      hw[(r + 8 * g) * HH + ncol] = (_Float16)sspf(acc[r] + bv);
  }
  __syncthreads();  // C-layout stores -> A-layout loads (cross-lane)

  // Reload staged activations as A fragments for GEMM2
  v16h a2frag[4];
#pragma unroll
  for (int kb = 0; kb < 4; ++kb) {
    v8h lo = *(const v8h*)&hw[m * HH + kb * 32 + 16 * g];
    v8h hi = *(const v8h*)&hw[m * HH + kb * 32 + 16 * g + 8];
    a2frag[kb] = __builtin_shufflevector(lo, hi, 0, 1, 2, 3, 4, 5, 6, 7, 8, 9,
                                         10, 11, 12, 13, 14, 15);
  }

  // GEMM2: [16,128]@[128,128] + bias, *gate, gather vlin[j], scatter agg[i]
#pragma unroll 1
  for (int nt = 0; nt < 8; ++nt) {
    v16h bfrag[4];
#pragma unroll
    for (int kb = 0; kb < 4; ++kb)
      bfrag[kb] = *(const v16h*)&b2[(nt * 16 + m) * HH + kb * 32 + 16 * g];
    sched_fence();
    v8f acc = {};
#pragma unroll
    for (int kb = 0; kb < 4; ++kb)
      acc = wmma_16x16x32(a2frag[kb], bfrag[kb], acc);

    const int ncol = nt * 16 + m;
    const float bv = sb2[ncol];
#pragma unroll
    for (int r = 0; r < 8; ++r) {
      const int er = r + 8 * g;  // local edge row
      const float w = (acc[r] + bv) * cg[wave][er];
      const float val = vlin[jr[wave][er] * HH + ncol] * w;
      atomicAdd(&agg[ir[wave][er] * HH + ncol], val);
    }
  }
}

// ---------------------------------------------------------------------------
// Readout tail: out[n,c] = sum_k h[n,k]*u2[k,c] + b2[c]   (c < 3)
// ---------------------------------------------------------------------------
__global__ void schnet_readout_k(const float* __restrict__ h,
                                 const float* __restrict__ u2w,
                                 const float* __restrict__ u2b,
                                 float* __restrict__ out) {
  int t = blockIdx.x * blockDim.x + threadIdx.x;
  if (t >= NN * 3) return;
  int n = t / 3, c = t % 3;
  float acc = u2b[c];
  const float* hr = h + n * HH;
#pragma unroll 4
  for (int k = 0; k < HH; ++k) acc += hr[k] * u2w[k * 3 + c];
  out[t] = acc;
}

// ---------------------------------------------------------------------------
extern "C" void kernel_launch(void* const* d_in, const int* in_sizes, int n_in,
                              void* d_out, int out_size, void* d_ws,
                              size_t ws_size, hipStream_t stream) {
  const float* z      = (const float*)d_in[0];
  const float* dist   = (const float*)d_in[1];
  const int*   eidx   = (const int*)d_in[2];
  const float* init_w = (const float*)d_in[3];
  const float* init_b = (const float*)d_in[4];

  const float* lin_w[2] = {(const float*)d_in[5], (const float*)d_in[14]};
  const float* m0w[2]   = {(const float*)d_in[6], (const float*)d_in[15]};
  const float* m0b[2]   = {(const float*)d_in[7], (const float*)d_in[16]};
  const float* m2w[2]   = {(const float*)d_in[8], (const float*)d_in[17]};
  const float* m2b[2]   = {(const float*)d_in[9], (const float*)d_in[18]};
  const float* l1w[2]   = {(const float*)d_in[10], (const float*)d_in[19]};
  const float* l1b[2]   = {(const float*)d_in[11], (const float*)d_in[20]};
  const float* l2w[2]   = {(const float*)d_in[12], (const float*)d_in[21]};
  const float* l2b[2]   = {(const float*)d_in[13], (const float*)d_in[22]};
  const float* u1w      = (const float*)d_in[23];
  const float* u1b      = (const float*)d_in[24];
  const float* u2w      = (const float*)d_in[25];
  const float* u2b      = (const float*)d_in[26];

  float* v    = (float*)d_ws;          // [N,128]
  float* vlin = v + NN * HH;           // [N,128]
  float* agg  = vlin + NN * HH;        // [N,128]
  float* hbuf = agg + NN * HH;         // [N,128]

  const int elemBlocks = (NN * HH + 255) / 256;
  const int gemmBlocks = (NN / 16 + 3) / 4;  // 3125 tiles, 4 waves/block

  schnet_init_k<<<elemBlocks, 256, 0, stream>>>(z, init_w, init_b, v);

  for (int L = 0; L < 2; ++L) {
    schnet_node_gemm_k<<<gemmBlocks, 128, 0, stream>>>(
        v, lin_w[L], nullptr, nullptr, vlin, NN, 0);
    schnet_zero_k<<<elemBlocks, 256, 0, stream>>>(agg, NN * HH);
    schnet_edge_k<<<EE / 32, 64, 0, stream>>>(dist, eidx, m0w[L], m0b[L],
                                              m2w[L], m2b[L], vlin, agg);
    schnet_node_gemm_k<<<gemmBlocks, 128, 0, stream>>>(
        agg, l1w[L], l1b[L], nullptr, hbuf, NN, 1);
    schnet_node_gemm_k<<<gemmBlocks, 128, 0, stream>>>(
        hbuf, l2w[L], l2b[L], v, v, NN, 0);
  }

  schnet_node_gemm_k<<<gemmBlocks, 128, 0, stream>>>(v, u1w, u1b, nullptr,
                                                     hbuf, NN, 1);
  schnet_readout_k<<<(NN * 3 + 255) / 256, 256, 0, stream>>>(
      hbuf, u2w, u2b, (float*)d_out);
}
